// LossComputation_40733469835978
// MI455X (gfx1250) — compile-verified, matrix operand order checked by reading, not run on hardware
//
#include <hip/hip_runtime.h>
#include <hip/hip_bf16.h>
#include <math.h>
#include <stdint.h>

typedef __attribute__((ext_vector_type(16))) _Float16 v16h;
typedef __attribute__((ext_vector_type(8)))  _Float16 v8h;
typedef __attribute__((ext_vector_type(8)))  float    v8f;

#define Bsz    256
#define Dd     512
#define Pp     5
#define NCc    11003
#define NPAD   11008
#define NTILES 688      // NPAD / 16
#define SEGC   6
#define Hh     64
#define SCALEc 28.0f
#define ALPHAc 0.6f
#define BETAc  0.4f
#define SPc    10.0f
#define SNc    40.0f

// ---------------------------------------------------------------------------
// helpers
// ---------------------------------------------------------------------------
__device__ inline float softplusf(float x) {
    return fmaxf(x, 0.0f) + log1pf(__expf(-fabsf(x)));
}

// block-wide reduce-add then one atomic; requires blockDim.x == 256
__device__ inline void blockAtomicAdd256(float v, float* dst, float scale) {
    __shared__ float buf[256];
    int t = threadIdx.x;
    buf[t] = v;
    __syncthreads();
    for (int s = 128; s > 0; s >>= 1) {
        if (t < s) buf[t] += buf[t + s];
        __syncthreads();
    }
    if (t == 0) atomicAdd(dst, buf[0] * scale);
}

// 16x32 f16 fragment from a row-major source (leading dim ld), vectorized:
// lane L: row r = L&15, half h = L>>4 ; element e: K = 8h + (e&7) + 16*(e>>3)
// -> two contiguous 16B loads per lane.
__device__ inline v16h load_frag_rm(const _Float16* base, int ld) {
    int lane = threadIdx.x & 31;
    int r = lane & 15, h = lane >> 4;
    const _Float16* p = base + r * ld + 8 * h;
    v8h lo = *(const v8h*)p;
    v8h hi = *(const v8h*)(p + 16);
    return __builtin_shufflevector(lo, hi, 0, 1, 2, 3, 4, 5, 6, 7,
                                   8, 9, 10, 11, 12, 13, 14, 15);
}

// ---------------------------------------------------------------------------
// kernels
// ---------------------------------------------------------------------------
__global__ void zero_out_k(float* out) {
    if (threadIdx.x < 4) out[threadIdx.x] = 0.0f;
}

// L2-normalize rows of (nrows x 512); write f16 always, f32 optionally
__global__ void normalize_rows_k(const float* __restrict__ in, float* fout,
                                 _Float16* __restrict__ hout) {
    int r = blockIdx.x;
    int t = threadIdx.x; // blockDim = 128
    __shared__ float red[128];
    float ss = 0.0f;
    for (int k = t; k < Dd; k += 128) {
        float x = in[r * Dd + k];
        ss += x * x;
    }
    red[t] = ss;
    __syncthreads();
    for (int s = 64; s > 0; s >>= 1) {
        if (t < s) red[t] += red[t + s];
        __syncthreads();
    }
    float rn = rsqrtf(red[0]);
    for (int k = t; k < Dd; k += 128) {
        float x = in[r * Dd + k] * rn;
        if (fout) fout[r * Dd + k] = x;
        hout[r * Dd + k] = (_Float16)x;
    }
}

// per-column L2 norms of W (512 x NC)
__global__ void w_col_norms_k(const float* __restrict__ W, float* __restrict__ wnorm) {
    int n = blockIdx.x * 256 + threadIdx.x;
    if (n >= NPAD) return;
    if (n >= NCc) { wnorm[n] = 1.0f; return; }
    float ss = 0.0f;
    for (int k = 0; k < Dd; ++k) {
        float x = W[k * NCc + n];
        ss += x * x;
    }
    wnorm[n] = sqrtf(ss);
}

// Normalized W packed directly into WMMA B-fragment order:
// Wpack[((t*16 + kc)*32 + lane)*16 + e] with
//   n = t*16 + (lane&15), h = lane>>4, K = kc*32 + 8h + (e&7) + 16*(e>>3)
// so each lane's fragment is one contiguous 32-byte v16h.
__global__ void w_pack_k(const float* __restrict__ W, const float* __restrict__ wnorm,
                         _Float16* __restrict__ Wpack) {
    int idx = blockIdx.x * 256 + threadIdx.x; // < NTILES*16*32*16 = 512*NPAD
    int e    = idx & 15;
    int lane = (idx >> 4) & 31;
    int kc   = (idx >> 9) & 15;
    int t    = idx >> 13;
    int n = t * 16 + (lane & 15);
    int h = lane >> 4;
    int K = kc * 32 + 8 * h + (e & 7) + 16 * (e >> 3);
    float v = (n < NCc) ? W[K * NCc + n] / wnorm[n] : 0.0f;
    Wpack[idx] = (_Float16)v;
}

// fused: logits = SCALE*(emb_n @ Wn) with online logsumexp per row.
// grid 32 blocks (2 matrices x 16 row-stripes), 256 threads = 8 waves.
// A stripe staged in LDS; B fragments are contiguous v16h loads from Wpack.
__global__ void logits_lse_k(const _Float16* __restrict__ vnh, const _Float16* __restrict__ tnh,
                             const _Float16* __restrict__ Wpack, float* __restrict__ lse) {
    int g = blockIdx.x;
    int mat = g >> 4, stripe = g & 15;
    const _Float16* emb = mat ? tnh : vnh;
    int tid = threadIdx.x, wid = tid >> 5, lane = tid & 31;

    // stage the 16x512 f16 A stripe into LDS (16 KB)
    __shared__ _Float16 Ash[16 * Dd];
    {
        const v8h* src = (const v8h*)(emb + (size_t)(stripe * 16) * Dd);
        v8h* dst = (v8h*)Ash;
        for (int i = tid; i < (16 * Dd) / 8; i += 256) dst[i] = src[i];
    }
    __syncthreads();

    float m[8], s[8];
#pragma unroll
    for (int j = 0; j < 8; ++j) { m[j] = -INFINITY; s[j] = 0.0f; }

    for (int t = 0; t < NTILES / 8; ++t) {
        // barrier: keep the LDS A-fragment loads inside the loop (no LICM ->
        // no 128-VGPR hoisted array -> no scratch spills)
        asm volatile("" ::: "memory");

        int tile = t * 8 + wid;

        // unconditional prefetch of a later tile's packed block (clamped)
        int tn_ = tile + 8 < NTILES ? tile + 8 : NTILES - 1;
        __builtin_prefetch(Wpack + (size_t)tn_ * 8192 + lane * 256, 0, 3);

        const _Float16* wp = Wpack + (size_t)tile * 8192; // 16 kc-blocks of 512 f16

        // two independent WMMA accumulation chains (even/odd kc)
        v8f acc0 = {}, acc1 = {};
#pragma unroll
        for (int kc = 0; kc < 16; kc += 2) {
            v16h a0 = load_frag_rm(Ash + kc * 32, Dd);
            v16h b0 = *(const v16h*)(wp + (size_t)(kc * 32 + lane) * 16);
            acc0 = __builtin_amdgcn_wmma_f32_16x16x32_f16(
                false, a0, false, b0, (short)0, acc0, false, false);
            v16h a1 = load_frag_rm(Ash + (kc + 1) * 32, Dd);
            v16h b1 = *(const v16h*)(wp + (size_t)((kc + 1) * 32 + lane) * 16);
            acc1 = __builtin_amdgcn_wmma_f32_16x16x32_f16(
                false, a1, false, b1, (short)0, acc1, false, false);
        }
        v8f acc = acc0 + acc1;

        int col = tile * 16 + (lane & 15);
        if (col < NCc) {
#pragma unroll
            for (int j = 0; j < 8; ++j) {
                float x = SCALEc * acc[j];
                float nm = fmaxf(m[j], x);
                s[j] = s[j] * __expf(m[j] - nm) + __expf(x - nm);
                m[j] = nm;
            }
        }
    }

    // butterfly merge across the 16 lanes of each half (xor masks 8,4,2,1)
#pragma unroll
    for (int j = 0; j < 8; ++j) {
        for (int off = 8; off > 0; off >>= 1) {
            float om = __shfl_xor(m[j], off, 32);
            float os = __shfl_xor(s[j], off, 32);
            float nm = fmaxf(m[j], om);
            s[j] = s[j] * __expf(m[j] - nm) + os * __expf(om - nm);
            m[j] = nm;
        }
    }

    __shared__ float lm[8][16], lsd[8][16];
    if (lane == 0) {
#pragma unroll
        for (int j = 0; j < 8; ++j) { lm[wid][j] = m[j]; lsd[wid][j] = s[j]; }
    }
    if (lane == 16) {
#pragma unroll
        for (int j = 0; j < 8; ++j) { lm[wid][8 + j] = m[j]; lsd[wid][8 + j] = s[j]; }
    }
    __syncthreads();
    if (tid < 16) {
        float M = -INFINITY, S = 0.0f;
        for (int w = 0; w < 8; ++w) {
            float om = lm[w][tid], os = lsd[w][tid];
            float nm = fmaxf(M, om);
            S = S * __expf(M - nm) + os * __expf(om - nm);
            M = nm;
        }
        lse[mat * Bsz + stripe * 16 + tid] = M + logf(S);
    }
}

// exact fp32 label logit + accumulate instance loss. grid 2 x 256 (one wave per row)
__global__ void label_loss_k(const float* __restrict__ vn, const float* __restrict__ tn,
                             const float* __restrict__ W, const float* __restrict__ wnorm,
                             const int* __restrict__ labels, const float* __restrict__ lse,
                             float* out) {
    int gw = (blockIdx.x * 256 + threadIdx.x) >> 5; // 0..15
    int lane = threadIdx.x & 31;
    int mat = gw >> 8, r = gw & 255;
    const float* e = (mat ? tn : vn) + r * Dd;
    int lab = labels[r];
    float sum = 0.0f;
    for (int k = lane; k < Dd; k += 32) sum += e[k] * W[k * NCc + lab];
    for (int off = 16; off > 0; off >>= 1) sum += __shfl_xor(sum, off, 32);
    if (lane == 0) {
        float logit = SCALEc * sum / wnorm[lab];
        atomicAdd(out + 0, (lse[gw] - logit) * (1.0f / (float)Bsz));
    }
}

// mask (segmentation CE) loss. grid 20480 x 256, one thread per pixel
__global__ void mask_loss_k(const float* __restrict__ seg, const int* __restrict__ masks,
                            float* out) {
    int pix = blockIdx.x * 256 + threadIdx.x; // < 1280*4096
    int img = pix >> 12, pp = pix & 4095;
    const float* base = seg + (size_t)img * SEGC * 4096 + pp;
    float x[SEGC];
    float mx = -INFINITY;
#pragma unroll
    for (int c = 0; c < SEGC; ++c) { x[c] = base[c * 4096]; mx = fmaxf(mx, x[c]); }
    float se = 0.0f;
#pragma unroll
    for (int c = 0; c < SEGC; ++c) se += __expf(x[c] - mx);
    float lse = mx + logf(se);
    int mk = masks[pix];
    float ce = lse - x[mk];
    blockAtomicAdd256(ce, out + 1, (float)Pp / (1280.0f * 4096.0f));
}

// global align: sim tile via WMMA fused with softplus reduction. grid 32 x 256
__global__ void global_align_k(const _Float16* __restrict__ vnh, const _Float16* __restrict__ tnh,
                               const int* __restrict__ labels, float* out) {
    int tid = threadIdx.x, wid = tid >> 5, lane = tid & 31;
    int tile = blockIdx.x * 8 + wid; // 0..255
    int tr = tile >> 4, tc = tile & 15;
    v8f acc0 = {}, acc1 = {};
#pragma unroll
    for (int kc = 0; kc < 16; kc += 2) {
        v16h a0 = load_frag_rm(vnh + (size_t)(tr * 16) * Dd + kc * 32, Dd);
        v16h b0 = load_frag_rm(tnh + (size_t)(tc * 16) * Dd + kc * 32, Dd);
        acc0 = __builtin_amdgcn_wmma_f32_16x16x32_f16(
            false, a0, false, b0, (short)0, acc0, false, false);
        v16h a1 = load_frag_rm(vnh + (size_t)(tr * 16) * Dd + (kc + 1) * 32, Dd);
        v16h b1 = load_frag_rm(tnh + (size_t)(tc * 16) * Dd + (kc + 1) * 32, Dd);
        acc1 = __builtin_amdgcn_wmma_f32_16x16x32_f16(
            false, a1, false, b1, (short)0, acc1, false, false);
    }
    v8f acc = acc0 + acc1;
    int h = lane >> 4, n = lane & 15;
    float local = 0.0f;
#pragma unroll
    for (int j = 0; j < 8; ++j) {
        int r = tr * 16 + j + 8 * h;
        int c = tc * 16 + n;
        float sim = acc[j];
        bool match = labels[r] == labels[c];
        local += match ? softplusf(-SPc * (sim - ALPHAc))
                       : softplusf( SNc * (sim - BETAc));
    }
    blockAtomicAdd256(local, out + 2, 2.0f / (float)Bsz);
}

// per-part similarity matrices (stored). grid 160 x 256
__global__ void part_sims_k(const _Float16* __restrict__ peh, const _Float16* __restrict__ aeh,
                            float* __restrict__ sims) {
    int tid = threadIdx.x, wid = tid >> 5, lane = tid & 31;
    int p = blockIdx.x >> 5;                  // 5 parts, 32 blocks each
    int tile = (blockIdx.x & 31) * 8 + wid;   // 0..255
    int tr = tile >> 4, tc = tile & 15;
    const _Float16* A = peh + (size_t)(p * Bsz + tr * 16) * Dd;
    const _Float16* Bm = aeh + (size_t)(p * Bsz + tc * 16) * Dd;
    v8f acc0 = {}, acc1 = {};
#pragma unroll
    for (int kc = 0; kc < 16; kc += 2) {
        v16h a0 = load_frag_rm(A + kc * 32, Dd);
        v16h b0 = load_frag_rm(Bm + kc * 32, Dd);
        acc0 = __builtin_amdgcn_wmma_f32_16x16x32_f16(
            false, a0, false, b0, (short)0, acc0, false, false);
        v16h a1 = load_frag_rm(A + (kc + 1) * 32, Dd);
        v16h b1 = load_frag_rm(Bm + (kc + 1) * 32, Dd);
        acc1 = __builtin_amdgcn_wmma_f32_16x16x32_f16(
            false, a1, false, b1, (short)0, acc1, false, false);
    }
    v8f acc = acc0 + acc1;
    int h = lane >> 4, n = lane & 15;
#pragma unroll
    for (int j = 0; j < 8; ++j) {
        int r = tr * 16 + j + 8 * h;
        int c = tc * 16 + n;
        sims[((size_t)p * Bsz + r) * Bsz + c] = acc[j];
    }
}

// top-8 per row (mode 0) and per column (mode 1). grid 10 x 256
__global__ void top8_k(const float* __restrict__ sims, int* __restrict__ rowtop,
                       int* __restrict__ coltop) {
    int gid = blockIdx.x * 256 + threadIdx.x;
    if (gid >= 2 * Pp * Bsz) return;
    int mode = gid / (Pp * Bsz);
    int rest = gid % (Pp * Bsz);
    int p = rest >> 8, r = rest & 255;
    const float* base = sims + (size_t)p * Bsz * Bsz;
    int chosen[8];
#pragma unroll 1
    for (int pass = 0; pass < 8; ++pass) {
        float bv = -INFINITY;
        int bi = -1;
        for (int c = 0; c < Bsz; ++c) {
            bool used = false;
            for (int q = 0; q < pass; ++q) used |= (chosen[q] == c);
            if (used) continue;
            float v = (mode == 0) ? base[r * Bsz + c] : base[c * Bsz + r];
            if (v > bv) { bv = v; bi = c; }
        }
        chosen[pass] = bi;
        int* dst = (mode == 0) ? rowtop : coltop;
        dst[(p * Bsz + r) * 8 + pass] = bi;
    }
}

// boost flags per part. grid 5 x 64
__global__ void boosts_k(const int* __restrict__ rowtop, const int* __restrict__ coltop,
                         int* __restrict__ boost1, int* __restrict__ boost2) {
    int p = blockIdx.x, t = threadIdx.x;
    for (int i = t; i < Bsz; i += 64) {
        boost1[p * Bsz + i] = 0;
        boost2[p * Bsz + i] = 0;
    }
    __syncthreads();
    if (t < 8) {
        int f = rowtop[(p * Bsz + p) * 8 + t];          // fwd1[j]
        int hit = 0;
        for (int k = 0; k < 8; ++k) hit |= (coltop[(p * Bsz + f) * 8 + k] == p);
        boost1[p * Bsz + f] = hit;
    } else if (t < 16) {
        int j = t - 8;
        int f = coltop[(p * Bsz + p) * 8 + j];          // fwd2[j]
        int hit = 0;
        for (int k = 0; k < 8; ++k) hit |= (rowtop[(p * Bsz + f) * 8 + k] == p);
        boost2[p * Bsz + f] = hit;
    }
}

// final local-align weighted sum. grid 1280 x 256, one thread per (p,r,c)
__global__ void local_final_k(const float* __restrict__ sims, const int* __restrict__ labels,
                              const unsigned char* __restrict__ vmask,
                              const unsigned char* __restrict__ tmask,
                              const int* __restrict__ boost1, const int* __restrict__ boost2,
                              float* out) {
    int idx = blockIdx.x * 256 + threadIdx.x; // < 5*256*256
    int p = idx >> 16;
    int rc = idx & 65535;
    int r = rc >> 8, c = rc & 255;
    float s = sims[idx];
    float Lp = softplusf(-SPc * (s - ALPHAc));
    float Ln = softplusf( SNc * (s - BETAc));
    bool match = labels[r] == labels[c];
    bool pm_r = vmask[r * Pp + p] != 0;
    bool pm_c = vmask[c * Pp + p] != 0;
    bool am_c = tmask[c * Pp + p] != 0;
    float val = 0.0f;
    if (pm_r && am_c) val += (match || boost1[p * Bsz + c]) ? Lp : Ln;
    if (pm_c && am_c && pm_r) val += (match || boost2[p * Bsz + r]) ? Lp : Ln;
    blockAtomicAdd256(val, out + 3, 1.0f / ((float)Bsz * (float)Pp));
}

// ---------------------------------------------------------------------------
// launch
// ---------------------------------------------------------------------------
extern "C" void kernel_launch(void* const* d_in, const int* in_sizes, int n_in,
                              void* d_out, int out_size, void* d_ws, size_t ws_size,
                              hipStream_t stream) {
    const float* visual  = (const float*)d_in[0];
    const float* textual = (const float*)d_in[1];
    const float* part    = (const float*)d_in[2];
    const float* attr    = (const float*)d_in[3];
    const float* seg     = (const float*)d_in[4];
    const float* W       = (const float*)d_in[5];
    const int*   labels  = (const int*)d_in[6];
    const int*   masks   = (const int*)d_in[7];
    const unsigned char* vmask = (const unsigned char*)d_in[8];
    const unsigned char* tmask = (const unsigned char*)d_in[9];
    float* out = (float*)d_out;

    // workspace carve-out
    uint8_t* ws = (uint8_t*)d_ws;
    size_t off = 0;
    auto alloc = [&](size_t bytes) -> void* {
        void* p = ws + off;
        off += (bytes + 255) & ~(size_t)255;
        return p;
    };
    float*    vn     = (float*)alloc((size_t)Bsz * Dd * 4);
    float*    tn     = (float*)alloc((size_t)Bsz * Dd * 4);
    _Float16* vnh    = (_Float16*)alloc((size_t)Bsz * Dd * 2);
    _Float16* tnh    = (_Float16*)alloc((size_t)Bsz * Dd * 2);
    _Float16* peh    = (_Float16*)alloc((size_t)Pp * Bsz * Dd * 2);
    _Float16* aeh    = (_Float16*)alloc((size_t)Pp * Bsz * Dd * 2);
    float*    wnorm  = (float*)alloc((size_t)NPAD * 4);
    _Float16* Wpack  = (_Float16*)alloc((size_t)Dd * NPAD * 2);
    float*    lse    = (float*)alloc((size_t)2 * Bsz * 4);
    float*    sims   = (float*)alloc((size_t)Pp * Bsz * Bsz * 4);
    int*      rowtop = (int*)alloc((size_t)Pp * Bsz * 8 * 4);
    int*      coltop = (int*)alloc((size_t)Pp * Bsz * 8 * 4);
    int*      boost1 = (int*)alloc((size_t)Pp * Bsz * 4);
    int*      boost2 = (int*)alloc((size_t)Pp * Bsz * 4);

    zero_out_k<<<1, 32, 0, stream>>>(out);

    // normalization (f32 + f16 copies)
    normalize_rows_k<<<Bsz, 128, 0, stream>>>(visual,  vn, vnh);
    normalize_rows_k<<<Bsz, 128, 0, stream>>>(textual, tn, tnh);
    normalize_rows_k<<<Pp * Bsz, 128, 0, stream>>>(part, nullptr, peh);
    normalize_rows_k<<<Pp * Bsz, 128, 0, stream>>>(attr, nullptr, aeh);

    // W column norms + fragment-packed normalized f16 copy
    w_col_norms_k<<<(NPAD + 255) / 256, 256, 0, stream>>>(W, wnorm);
    w_pack_k<<<(Dd * NPAD) / 256, 256, 0, stream>>>(W, wnorm, Wpack);

    // instance loss: WMMA logits scan (logsumexp) + exact label logits
    logits_lse_k<<<32, 256, 0, stream>>>(vnh, tnh, Wpack, lse);
    label_loss_k<<<2, 256, 0, stream>>>(vn, tn, W, wnorm, labels, lse, out);

    // mask loss
    mask_loss_k<<<(1280 * 4096) / 256, 256, 0, stream>>>(seg, masks, out);

    // global align loss (fused WMMA + softplus reduction)
    global_align_k<<<32, 256, 0, stream>>>(vnh, tnh, labels, out);

    // local align loss
    part_sims_k<<<Pp * 32, 256, 0, stream>>>(peh, aeh, sims);
    top8_k<<<(2 * Pp * Bsz + 255) / 256, 256, 0, stream>>>(sims, rowtop, coltop);
    boosts_k<<<Pp, 64, 0, stream>>>(rowtop, coltop, boost1, boost2);
    local_final_k<<<Pp * Bsz, 256, 0, stream>>>(sims, labels, vmask, tmask,
                                                boost1, boost2, out);

    (void)in_sizes; (void)n_in; (void)out_size; (void)ws_size;
}
